// NorseIFNet_66297115181590
// MI455X (gfx1250) — matrix operand order repaired and checked by reading, not compiled, not run on hardware
//
#include <hip/hip_runtime.h>

typedef __attribute__((ext_vector_type(16))) _Float16 v16h;
typedef __attribute__((ext_vector_type(8)))  _Float16 v8h;
typedef __attribute__((ext_vector_type(8)))  float    v8f;
typedef __attribute__((ext_vector_type(4)))  float    v4f;

#define BATCH   256
#define T_STEPS 128
#define DIN     784
#define DINP    800      // DIN padded to multiple of 32 (K granularity of WMMA f16)
#define H       2048
#define OUTN    10
#define OUTP    16

#define COLSTR  40       // LDS column stride in halves (80B) -> conflict-free banks

// ---- workspace layout (bytes) ----
#define OFF_W0H 0
#define OFF_W1H (OFF_W0H + (size_t)H * DINP * 2)     // f16 W0 [2048][800] (padded)
#define OFF_W2H (OFF_W1H + (size_t)H * H * 2)        // f16 W1 [2048][2048]
#define OFF_V0  (OFF_W2H + (size_t)OUTP * H * 2)     // f16 W2 [16][2048]  (padded rows = 0)
#define OFF_V1  (OFF_V0  + (size_t)H * BATCH * 4)
#define OFF_V2  (OFF_V1  + (size_t)H * BATCH * 4)
#define OFF_Z0  (OFF_V2  + (size_t)OUTP * BATCH * 4)
#define OFF_Z1  (OFF_Z0  + (size_t)BATCH * H * 2)

// One-time prep: fp32 -> f16 weights (+padding), zero membranes, out = t_max.
__global__ __launch_bounds__(256)
void prep_kernel(const float* __restrict__ W0, const float* __restrict__ W1,
                 const float* __restrict__ W2,
                 _Float16* __restrict__ W0h, _Float16* __restrict__ W1h,
                 _Float16* __restrict__ W2h,
                 float* __restrict__ V0, float* __restrict__ V1,
                 float* __restrict__ V2, float* __restrict__ Out)
{
    size_t i = (size_t)blockIdx.x * 256 + threadIdx.x;
    if (i < (size_t)H * H) W1h[i] = (_Float16)W1[i];
    if (i < (size_t)H * DINP) {
        size_t r = i / DINP, c = i % DINP;
        W0h[i] = (c < DIN) ? (_Float16)W0[r * DIN + c] : (_Float16)0.0f;
    }
    if (i < (size_t)OUTP * H)
        W2h[i] = (i < (size_t)OUTN * H) ? (_Float16)W2[i] : (_Float16)0.0f;
    if (i < (size_t)H * BATCH) { V0[i] = 0.0f; V1[i] = 0.0f; }
    if (i < (size_t)OUTP * BATCH) V2[i] = 0.0f;
    if (i < (size_t)BATCH * OUTN) Out[i] = 256.0f;   // t_max
}

__device__ __forceinline__ v16h cat8(v8h a, v8h b) {
    return __builtin_shufflevector(a, b, 0,1,2,3,4,5,6,7,8,9,10,11,12,13,14,15);
}

// IF epilogue shared by the hidden-layer kernels.
__device__ __forceinline__
void if_epilogue(v8f* acc, float* __restrict__ V, _Float16* __restrict__ Z,
                 int mtile, int ngroup, int nbase, int mbase,
                 int lane, int ln, int hi, float vth)
{
#pragma unroll
    for (int j = 0; j < 4; ++j) {
        const int n = nbase + j * 16 + ln;
#pragma unroll
        for (int r = 0; r < 8; ++r) {
            float vv = acc[j][r];
            bool  zb = vv > vth;
            acc[j][r] = zb ? 0.0f : vv;
            const int m = mbase + r + 8 * hi;
            Z[(size_t)n * H + m] = zb ? (_Float16)1.0f : (_Float16)0.0f;
        }
        const int tile = mtile * 16 + ngroup * 4 + j;
        *(v8f*)(V + (size_t)tile * 256 + lane * 8) = acc[j];
    }
}

// ---------------- Layer 1: z0(f16) @ W1^T, B tile staged via async-to-LDS ----
// Grid: 128 blocks x 128 thr. Block = 4 waves sharing one 64-batch-column B
// tile (double-buffered in LDS, filled with global_load_async_to_lds_b128,
// pipelined with s_wait_asynccnt). Wave w -> 16(M) x 64(N) output block.
__global__ __launch_bounds__(128)
void gemm_l1_kernel(const _Float16* __restrict__ A,   // W1h [2048][2048]
                    const _Float16* __restrict__ Bz,  // z0  [256][2048]
                    float* __restrict__ V,
                    _Float16* __restrict__ Z,
                    float vth)
{
    __shared__ _Float16 shB[2][64 * COLSTR];

    const int tid  = threadIdx.x;
    const int lane = tid & 31;
    const int hi   = lane >> 4;
    const int ln   = lane & 15;
    const int w    = tid >> 5;
    const int mgroup = blockIdx.x >> 2;
    const int ngroup = blockIdx.x & 3;
    const int mtile  = mgroup * 4 + w;
    const int mbase  = mtile * 16;
    const int nbase  = ngroup * 64;

    // Fill roles: 256 16B-chunks per tile (64 cols x 4), 2 per thread.
    const int c0 = tid * 2;
    const int colA = c0 >> 2,       qA = c0 & 3;
    const int colB = (c0 + 1) >> 2, qB = (c0 + 1) & 3;
    const _Float16* sA = Bz + (size_t)(nbase + colA) * H + qA * 8;
    const _Float16* sB = Bz + (size_t)(nbase + colB) * H + qB * 8;
    const unsigned dA = (unsigned)(size_t)&shB[0][colA * COLSTR + qA * 8];
    const unsigned dB = (unsigned)(size_t)&shB[0][colB * COLSTR + qB * 8];
    const unsigned bufstr = 64 * COLSTR * 2;   // bytes per buffer

    auto fill = [&](int buf, int k0) {
        const _Float16* s0 = sA + k0;
        const _Float16* s1 = sB + k0;
        unsigned d0 = dA + (unsigned)buf * bufstr;
        unsigned d1 = dB + (unsigned)buf * bufstr;
        asm volatile("global_load_async_to_lds_b128 %0, %1, off"
                     :: "v"(d0), "v"(s0) : "memory");
        asm volatile("global_load_async_to_lds_b128 %0, %1, off"
                     :: "v"(d1), "v"(s1) : "memory");
    };

    v8f acc[4];
#pragma unroll
    for (int j = 0; j < 4; ++j) {
        const int tile = mtile * 16 + ngroup * 4 + j;
        acc[j] = *(const v8f*)(V + (size_t)tile * 256 + lane * 8);
    }

    const _Float16* Arow = A + (size_t)(mbase + ln) * H;

    fill(0, 0);
    for (int it = 0; it < 64; ++it) {
        const int k0  = it * 32;
        const int cur = it & 1;
        // Unconditional wrap-around prefetch keeps wait math branch-free;
        // the final (wrapped) fill is never read.
        fill(cur ^ 1, ((it + 1) & 63) * 32);
        // Async loads complete in order: <=2 outstanding retires cur's pair.
        asm volatile("s_wait_asynccnt 0x2" ::: "memory");
        __syncthreads();

        v8h a0 = *(const v8h*)(Arow + k0 + 16 * hi);
        v8h a1 = *(const v8h*)(Arow + k0 + 16 * hi + 8);
        v16h a = cat8(a0, a1);
#pragma unroll
        for (int j = 0; j < 4; ++j) {
            const _Float16* bp = &shB[cur][(j * 16 + ln) * COLSTR + hi * 16];
            v16h b = cat8(*(const v8h*)bp, *(const v8h*)(bp + 8));
            acc[j] = __builtin_amdgcn_wmma_f32_16x16x32_f16(
                         false, a, false, b, (short)0, acc[j], false, false);
        }
        __syncthreads();   // all waves done reading before next overwrite
    }

    if_epilogue(acc, V, Z, mtile, ngroup, nbase, mbase, lane, ln, hi, vth);
}

// ---------------- Layer 0: x_t(f32) @ W0^T, B converted f32->f16 into LDS ----
__global__ __launch_bounds__(128)
void gemm_l0_kernel(const _Float16* __restrict__ A,  // W0h [2048][800]
                    const float* __restrict__ Xf,    // x (B,T,DIN) f32
                    int t,
                    float* __restrict__ V,
                    _Float16* __restrict__ Z,
                    float vth)
{
    __shared__ _Float16 shB[2][64 * COLSTR];

    const int tid  = threadIdx.x;
    const int lane = tid & 31;
    const int hi   = lane >> 4;
    const int ln   = lane & 15;
    const int w    = tid >> 5;
    const int mgroup = blockIdx.x >> 2;
    const int ngroup = blockIdx.x & 3;
    const int mtile  = mgroup * 4 + w;
    const int mbase  = mtile * 16;
    const int nbase  = ngroup * 64;

    // Fill role: thread -> (col, 16-wide K half)
    const int col = tid >> 1, h2 = tid & 1;
    const float* xrow = Xf + ((size_t)(nbase + col) * T_STEPS + t) * DIN;
    _Float16* dst0 = &shB[0][col * COLSTR + h2 * 16];

    auto fill = [&](int buf, int k0) {
        const int kk = k0 + h2 * 16;
        v8h lo = {}, hh = {};
        if (kk < DIN) {          // DIN 16-aligned: chunk fully valid or fully pad
            const float* px = xrow + kk;
            v4f x0 = *(const v4f*)(px + 0);
            v4f x1 = *(const v4f*)(px + 4);
            v4f x2 = *(const v4f*)(px + 8);
            v4f x3 = *(const v4f*)(px + 12);
#pragma unroll
            for (int i = 0; i < 4; ++i) {
                lo[i]     = (_Float16)x0[i];
                lo[4 + i] = (_Float16)x1[i];
                hh[i]     = (_Float16)x2[i];
                hh[4 + i] = (_Float16)x3[i];
            }
        }
        _Float16* d = dst0 + buf * (64 * COLSTR);
        *(v8h*)d       = lo;
        *(v8h*)(d + 8) = hh;
    };

    v8f acc[4];
#pragma unroll
    for (int j = 0; j < 4; ++j) {
        const int tile = mtile * 16 + ngroup * 4 + j;
        acc[j] = *(const v8f*)(V + (size_t)tile * 256 + lane * 8);
    }

    const _Float16* Arow = A + (size_t)(mbase + ln) * DINP;

    fill(0, 0);
    for (int it = 0; it < 25; ++it) {            // 25 * 32 = 800 = DINP
        const int k0  = it * 32;
        const int cur = it & 1;
        if (it + 1 < 25) fill(cur ^ 1, k0 + 32);
        __syncthreads();

        v8h a0 = *(const v8h*)(Arow + k0 + 16 * hi);
        v8h a1 = *(const v8h*)(Arow + k0 + 16 * hi + 8);
        v16h a = cat8(a0, a1);
#pragma unroll
        for (int j = 0; j < 4; ++j) {
            const _Float16* bp = &shB[cur][(j * 16 + ln) * COLSTR + hi * 16];
            v16h b = cat8(*(const v8h*)bp, *(const v8h*)(bp + 8));
            acc[j] = __builtin_amdgcn_wmma_f32_16x16x32_f16(
                         false, a, false, b, (short)0, acc[j], false, false);
        }
        __syncthreads();
    }

    if_epilogue(acc, V, Z, mtile, ngroup, nbase, mbase, lane, ln, hi, vth);
}

// ---------------- Layer 2: M=16 (10 real), N=256, K=2048; first-spike times --
__global__ __launch_bounds__(128)
void gemm_out_kernel(const _Float16* __restrict__ A,   // W2h [16][2048]
                     const _Float16* __restrict__ Bz,  // z1  [256][2048]
                     float* __restrict__ V,            // v2 tile-blocked [16*256]
                     float* __restrict__ Out,          // [256][10]
                     int t)
{
    const int lane = threadIdx.x & 31;
    const int hi   = lane >> 4;
    const int ln   = lane & 15;
    const int ngroup = threadIdx.x >> 5;   // 0..3
    const int nbase  = ngroup * 64;

    v8f acc[4];
#pragma unroll
    for (int j = 0; j < 4; ++j)
        acc[j] = *(const v8f*)(V + (size_t)(ngroup * 4 + j) * 256 + lane * 8);

    const _Float16* Arow = A + (size_t)ln * H;

    for (int k0 = 0; k0 < H; k0 += 32) {
        v8h a0 = *(const v8h*)(Arow + k0 + 16 * hi);
        v8h a1 = *(const v8h*)(Arow + k0 + 16 * hi + 8);
        v16h a = cat8(a0, a1);
        const int kk = k0 + 16 * hi;
        if (k0 + 32 < H) __builtin_prefetch(Arow + k0 + 32, 0, 1);
#pragma unroll
        for (int j = 0; j < 4; ++j) {
            const int n = nbase + j * 16 + ln;
            v16h b = *(const v16h*)(Bz + (size_t)n * H + kk);
            acc[j] = __builtin_amdgcn_wmma_f32_16x16x32_f16(
                         false, a, false, b, (short)0, acc[j], false, false);
        }
    }

#pragma unroll
    for (int j = 0; j < 4; ++j) {
        const int n = nbase + j * 16 + ln;
#pragma unroll
        for (int r = 0; r < 8; ++r) {
            float vv = acc[j][r];
            bool  zb = vv > 1.0f;               // VTH layer 2
            acc[j][r] = zb ? 0.0f : vv;
            const int m = r + 8 * hi;
            if (m < OUTN && zb) {
                float* po = Out + (size_t)n * OUTN + m;
                if (*po == 256.0f) *po = (float)t;   // first spike only
            }
        }
        *(v8f*)(V + (size_t)(ngroup * 4 + j) * 256 + lane * 8) = acc[j];
    }
}

extern "C" void kernel_launch(void* const* d_in, const int* in_sizes, int n_in,
                              void* d_out, int out_size, void* d_ws, size_t ws_size,
                              hipStream_t stream)
{
    (void)in_sizes; (void)n_in; (void)out_size; (void)ws_size;

    const float* x  = (const float*)d_in[0];
    const float* W0 = (const float*)d_in[1];
    const float* W1 = (const float*)d_in[2];
    const float* W2 = (const float*)d_in[3];
    float* out = (float*)d_out;

    char* ws = (char*)d_ws;
    _Float16* W0h = (_Float16*)(ws + OFF_W0H);
    _Float16* W1h = (_Float16*)(ws + OFF_W1H);
    _Float16* W2h = (_Float16*)(ws + OFF_W2H);
    float*    V0  = (float*)(ws + OFF_V0);
    float*    V1  = (float*)(ws + OFF_V1);
    float*    V2  = (float*)(ws + OFF_V2);
    _Float16* Z0  = (_Float16*)(ws + OFF_Z0);
    _Float16* Z1  = (_Float16*)(ws + OFF_Z1);

    prep_kernel<<<((size_t)H * H + 255) / 256, 256, 0, stream>>>(
        W0, W1, W2, W0h, W1h, W2h, V0, V1, V2, out);

    for (int t = 0; t < T_STEPS; ++t) {
        gemm_l0_kernel<<<128, 128, 0, stream>>>(W0h, x, t, V0, Z0, 10.0f);
        gemm_l1_kernel<<<128, 128, 0, stream>>>(W1h, Z0, V1, Z1, 5.0f);
        gemm_out_kernel<<<1, 128, 0, stream>>>(W2h, Z1, V2, out, t);
    }
}
